// PAM_Module_75642964017282
// MI455X (gfx1250) — matrix operand order repaired and checked by reading, not compiled, not run on hardware
//
#include <hip/hip_runtime.h>

// ---------------------------------------------------------------------------
// PAM attention for MI455X (gfx1250): all matmuls via v_wmma_f32_16x16x32_bf16
// B=4, C=512, H=W=64 -> N=4096, d=C/8=64
// ---------------------------------------------------------------------------

typedef __bf16 bf16;
typedef __attribute__((ext_vector_type(16))) __bf16 v16bf;
typedef __attribute__((ext_vector_type(8)))  __bf16 v8bf;
typedef __attribute__((ext_vector_type(8)))  float  v8f;

union AB16 { v16bf v; v8bf h[2]; };

#define WMMA_BF16(A, Bm, Cm) \
  __builtin_amdgcn_wmma_f32_16x16x32_bf16(false, (A), false, (Bm), (short)0, (Cm), false, false)

static constexpr int BATCH = 4;
static constexpr int C     = 512;
static constexpr int C8    = 64;
static constexpr int N     = 4096;

__device__ __forceinline__ float wave_max(float v) {
  #pragma unroll
  for (int o = 16; o > 0; o >>= 1) v = fmaxf(v, __shfl_xor(v, o, 32));
  return v;
}
__device__ __forceinline__ float wave_sum(float v) {
  #pragma unroll
  for (int o = 16; o > 0; o >>= 1) v += __shfl_xor(v, o, 32);
  return v;
}

// ---------------------------------------------------------------------------
// Kernel 1: fp32 -> bf16 elementwise (weights)
// ---------------------------------------------------------------------------
__global__ void cvt_bf16_kernel(const float* __restrict__ src, bf16* __restrict__ dst, int n) {
  int i = blockIdx.x * 256 + threadIdx.x;
  if (i < n) dst[i] = (bf16)src[i];
}

// ---------------------------------------------------------------------------
// Kernel 2: x [B,C,N] f32 -> xT [B,N,C] bf16, 32x32 LDS-tiled transpose
// grid (N/32, C/32, B), block (32, 8)
// ---------------------------------------------------------------------------
__global__ void xpose_kernel(const float* __restrict__ x, bf16* __restrict__ xT) {
  __shared__ float t[32][33];
  int b = blockIdx.z, c0 = blockIdx.y * 32, n0 = blockIdx.x * 32;
  int tx = threadIdx.x, ty = threadIdx.y;
  const float* xp = x + ((size_t)b * C + c0) * N + n0;
  #pragma unroll
  for (int k = 0; k < 32; k += 8)
    t[ty + k][tx] = xp[(size_t)(ty + k) * N + tx];
  __syncthreads();
  bf16* op = xT + ((size_t)b * N + n0) * C + c0;
  #pragma unroll
  for (int k = 0; k < 32; k += 8)
    op[(size_t)(ty + k) * C + tx] = (bf16)t[tx][ty + k];
}

// ---------------------------------------------------------------------------
// Kernel 3: q/k projections -> qT,kT [B,N,64] bf16
// grid (N/16, B), block 256. Wave w: sel = w>>2 (0:q, 1:k), o-tile = w&3.
// D(m=i, n=o) = sum_c xT(i,c) * W(o,c)
// ---------------------------------------------------------------------------
__global__ void __launch_bounds__(256)
proj_qk_kernel(const bf16* __restrict__ xT,
               const bf16* __restrict__ Wqb, const bf16* __restrict__ Wkb,
               const float* __restrict__ bq, const float* __restrict__ bk,
               bf16* __restrict__ qT, bf16* __restrict__ kT) {
  int b = blockIdx.y;
  int ibase = blockIdx.x * 16;
  int w = threadIdx.x >> 5, l = threadIdx.x & 31;
  int sel = w >> 2;
  int obase = (w & 3) * 16;
  const bf16*  Wb   = sel ? Wkb : Wqb;
  const float* bias = sel ? bk  : bq;
  bf16*        outT = sel ? kT  : qT;

  const bf16* xrow = xT + ((size_t)b * N + ibase + (l & 15)) * C;   // A: row i
  int k0 = (l >> 4) * 16;
  const bf16* wrow = Wb + (size_t)(obase + (l & 15)) * C + (l >> 4) * 16;  // B: col o

  v8f acc = {};
  for (int ks = 0; ks < 16; ++ks) {           // K = C = 512 in steps of 32
    AB16 a;
    a.h[0] = *(const v8bf*)(xrow + ks * 32 + k0);
    a.h[1] = *(const v8bf*)(xrow + ks * 32 + k0 + 8);
    v16bf bfrag = *(const v16bf*)(wrow + ks * 32);
    acc = WMMA_BF16(a.v, bfrag, acc);
  }
  int o  = obase + (l & 15);
  float bo = bias[o];
  int mhi = (l >> 4) * 8;
  bf16* orow = outT + ((size_t)b * N + ibase) * C8 + o;
  #pragma unroll
  for (int r = 0; r < 8; ++r)
    orow[(size_t)(mhi + r) * C8] = (bf16)(acc[r] + bo);
}

// ---------------------------------------------------------------------------
// Kernel 4: v = Wv x + bv -> v [B,C,N] bf16
// grid (N/64, C/32, B), block 256. Wave w: o-sub = w>>2, i-sub = w&3.
// D(m=o, n=i) = sum_c Wv(o,c) * xT(i,c)
// ---------------------------------------------------------------------------
__global__ void __launch_bounds__(256)
proj_v_kernel(const bf16* __restrict__ xT, const bf16* __restrict__ Wvb,
              const float* __restrict__ bv, bf16* __restrict__ vout) {
  int b = blockIdx.z;
  int w = threadIdx.x >> 5, l = threadIdx.x & 31;
  int obase = blockIdx.y * 32 + (w >> 2) * 16;
  int ibase = blockIdx.x * 64 + (w & 3) * 16;
  const bf16* wrow = Wvb + (size_t)(obase + (l & 15)) * C;                 // A: row o
  int k0 = (l >> 4) * 16;
  const bf16* xrow = xT + ((size_t)b * N + ibase + (l & 15)) * C + (l >> 4) * 16;  // B: col i

  v8f acc = {};
  for (int ks = 0; ks < 16; ++ks) {
    AB16 a;
    a.h[0] = *(const v8bf*)(wrow + ks * 32 + k0);
    a.h[1] = *(const v8bf*)(wrow + ks * 32 + k0 + 8);
    v16bf bfrag = *(const v16bf*)(xrow + ks * 32);
    acc = WMMA_BF16(a.v, bfrag, acc);
  }
  int i   = ibase + (l & 15);
  int mhi = (l >> 4) * 8;
  bf16* op = vout + ((size_t)b * C + obase) * N + i;
  #pragma unroll
  for (int r = 0; r < 8; ++r)
    op[(size_t)(mhi + r) * N] = (bf16)(acc[r] + bv[obase + mhi + r]);
}

// ---------------------------------------------------------------------------
// Kernel 5: fused attention per (batch, 32-query block).
// Pass 1: energy[32][4096] bf16 -> 256 KB dynamic LDS (WMMA, K=64, 2 m-tiles)
// Pass 2: row softmax in place (fp32 max/sum via shuffles, p stored bf16)
// Pass 3: out[32 x 512] = p x v^T; each wave owns 64 channels; each B-fragment
//         feeds 2 WMMAs (both m-tiles) -> halved V traffic, no inner-loop cvt
// Epilogue: out = gamma * out * inv_sum + x   (residual)
// ---------------------------------------------------------------------------
__global__ void __launch_bounds__(256, 1)
attn_kernel(const bf16* __restrict__ qT, const bf16* __restrict__ kT,
            const bf16* __restrict__ vmat, const float* __restrict__ x,
            const float* __restrict__ gamma, float* __restrict__ out) {
  extern __shared__ char smem[];
  bf16*  eng = (bf16*)smem;                       // [32][N] bf16 logits -> probs
  float* inv = (float*)(smem + (size_t)32 * N * 2);  // [32] 1/rowsum

  int b = blockIdx.y;
  int ibase = blockIdx.x * 32;
  int w = threadIdx.x >> 5, l = threadIdx.x & 31;
  int mhi = (l >> 4) * 8;
  int k0  = (l >> 4) * 16;

  // ---- Pass 1: energy(i,j) = sum_c qT(i,c) kT(j,c); wave w owns j in [w*512, w*512+512)
  AB16 a0[2], a1[2];
  #pragma unroll
  for (int mt = 0; mt < 2; ++mt) {
    const bf16* qrow = qT + ((size_t)b * N + ibase + mt * 16 + (l & 15)) * C8;
    a0[mt].h[0] = *(const v8bf*)(qrow + k0);
    a0[mt].h[1] = *(const v8bf*)(qrow + k0 + 8);
    a1[mt].h[0] = *(const v8bf*)(qrow + 32 + k0);
    a1[mt].h[1] = *(const v8bf*)(qrow + 32 + k0 + 8);
  }

  for (int t = 0; t < 32; ++t) {
    int j = w * 512 + t * 16 + (l & 15);
    const bf16* krow = kT + ((size_t)b * N + j) * C8 + (l >> 4) * 16;
    v16bf b0 = *(const v16bf*)(krow);
    v16bf b1 = *(const v16bf*)(krow + 32);
    #pragma unroll
    for (int mt = 0; mt < 2; ++mt) {
      v8f acc = {};
      acc = WMMA_BF16(a0[mt].v, b0, acc);
      acc = WMMA_BF16(a1[mt].v, b1, acc);
      v8bf pk = __builtin_convertvector(acc, v8bf);   // packed f32->bf16
      #pragma unroll
      for (int r = 0; r < 8; ++r)
        eng[(size_t)(mt * 16 + mhi + r) * N + j] = pk[r];
    }
  }
  __syncthreads();

  // ---- Pass 2: row softmax (4 rows per wave), overwrite logits with exp(e - max)
  for (int r = w * 4; r < w * 4 + 4; ++r) {
    bf16* row = eng + (size_t)r * N;
    float mx = -3.4e38f;
    for (int j = l * 8; j < N; j += 256) {
      v8f fv = __builtin_convertvector(*(const v8bf*)(row + j), v8f);
      #pragma unroll
      for (int e = 0; e < 8; ++e) mx = fmaxf(mx, fv[e]);
    }
    mx = wave_max(mx);
    float s = 0.f;
    for (int j = l * 8; j < N; j += 256) {
      v8f fv = __builtin_convertvector(*(const v8bf*)(row + j), v8f);
      v8f pv;
      #pragma unroll
      for (int e = 0; e < 8; ++e) { pv[e] = __expf(fv[e] - mx); s += pv[e]; }
      *(v8bf*)(row + j) = __builtin_convertvector(pv, v8bf);
    }
    s = wave_sum(s);
    if (l == 0) inv[r] = 1.0f / s;
  }
  __syncthreads();

  // ---- Pass 3: out(i,c) = sum_j p(i,j) v(c,j); wave w owns channels [w*64, w*64+64)
  int cw = w * 64;
  v8f acc[2][4] = {};
  for (int js = 0; js < 128; ++js) {
    int jbase = js * 32;
    int j0 = jbase + k0;
    AB16 a[2];
    #pragma unroll
    for (int mt = 0; mt < 2; ++mt) {
      const bf16* prow = eng + (size_t)(mt * 16 + (l & 15)) * N;
      a[mt].h[0] = *(const v8bf*)(prow + j0);
      a[mt].h[1] = *(const v8bf*)(prow + j0 + 8);
    }
    #pragma unroll
    for (int t = 0; t < 4; ++t) {
      int c = cw + t * 16 + (l & 15);
      const bf16* vrow = vmat + ((size_t)b * C + c) * N + jbase + (l >> 4) * 16;
      v16bf bfrag = *(const v16bf*)vrow;
      #pragma unroll
      for (int mt = 0; mt < 2; ++mt)
        acc[mt][t] = WMMA_BF16(a[mt].v, bfrag, acc[mt][t]);
    }
  }

  // ---- Epilogue: gamma * out * inv_sum + residual
  float g = gamma[0];
  #pragma unroll
  for (int mt = 0; mt < 2; ++mt) {
    #pragma unroll
    for (int t = 0; t < 4; ++t) {
      int c = cw + t * 16 + (l & 15);
      const float* xrow = x   + ((size_t)b * C + c) * N + ibase + mt * 16;
      float*       orow = out + ((size_t)b * C + c) * N + ibase + mt * 16;
      #pragma unroll
      for (int r = 0; r < 8; ++r) {
        int m = mhi + r;
        orow[m] = g * acc[mt][t][r] * inv[mt * 16 + m] + xrow[m];
      }
    }
  }
}

// ---------------------------------------------------------------------------
// Host launcher
// ---------------------------------------------------------------------------
extern "C" void kernel_launch(void* const* d_in, const int* in_sizes, int n_in,
                              void* d_out, int out_size, void* d_ws, size_t ws_size,
                              hipStream_t stream) {
  const float* x     = (const float*)d_in[0];
  const float* Wq    = (const float*)d_in[1];
  const float* bq    = (const float*)d_in[2];
  const float* Wk    = (const float*)d_in[3];
  const float* bk    = (const float*)d_in[4];
  const float* Wv    = (const float*)d_in[5];
  const float* bv    = (const float*)d_in[6];
  const float* gamma = (const float*)d_in[7];
  float* out = (float*)d_out;

  // Workspace layout (bf16), region offsets
  char* ws = (char*)d_ws;
  size_t off = 0;
  bf16* xT  = (bf16*)(ws + off); off += (size_t)BATCH * N * C * 2;     // 16 MB
  bf16* vbf = (bf16*)(ws + off); off += (size_t)BATCH * C * N * 2;     // 16 MB
  bf16* qT  = (bf16*)(ws + off); off += (size_t)BATCH * N * C8 * 2;    // 2 MB
  bf16* kT  = (bf16*)(ws + off); off += (size_t)BATCH * N * C8 * 2;    // 2 MB
  bf16* Wqb = (bf16*)(ws + off); off += (size_t)C8 * C * 2;
  bf16* Wkb = (bf16*)(ws + off); off += (size_t)C8 * C * 2;
  bf16* Wvb = (bf16*)(ws + off); off += (size_t)C * C * 2;

  // 1. Weight conversion
  cvt_bf16_kernel<<<(C8 * C + 255) / 256, 256, 0, stream>>>(Wq, Wqb, C8 * C);
  cvt_bf16_kernel<<<(C8 * C + 255) / 256, 256, 0, stream>>>(Wk, Wkb, C8 * C);
  cvt_bf16_kernel<<<(C * C + 255) / 256, 256, 0, stream>>>(Wv, Wvb, C * C);

  // 2. Transpose x -> xT bf16
  xpose_kernel<<<dim3(N / 32, C / 32, BATCH), dim3(32, 8), 0, stream>>>(x, xT);

  // 3. Q/K projections (transposed layout)
  proj_qk_kernel<<<dim3(N / 16, BATCH), 256, 0, stream>>>(xT, Wqb, Wkb, bq, bk, qT, kT);

  // 4. V projection ([C,N] layout)
  proj_v_kernel<<<dim3(N / 64, C / 32, BATCH), 256, 0, stream>>>(xT, Wvb, bv, vbf);

  // 5. Fused attention: 32-query blocks, 256 KB bf16 stripe + 128 B inv-sums
  size_t shmem = (size_t)32 * N * 2 + 128;
  attn_kernel<<<dim3(N / 32, BATCH), 256, shmem, stream>>>(qT, kT, vbf, x, gamma, out);
}
